// SynthesisLayer_55370718380350
// MI455X (gfx1250) — matrix operand order repaired and checked by reading, not compiled, not run on hardware
//
#include <hip/hip_runtime.h>

// ---------------------------------------------------------------------------
// StyleGAN2 SynthesisLayer: modulated 3x3 conv, N=8, IC=OC=128, RES=256.
// Implicit-GEMM conv on CDNA5 WMMA bf16 (v_wmma_f32_16x16x32_bf16), f32 accum.
// Conv kernel: halo slab of x staged once in LDS as bf16; barrier-free main
// loop of ds_load_b128 (A frags, immediate offsets) + global_load_b128
// (B frags, L2-resident weights) + 8 WMMAs per K-chunk per wave.
// ---------------------------------------------------------------------------

#define N_SMP 8
#define IC    128
#define OC    128
#define RES   256
#define WDIM  512
#define KTOT  (9 * IC)        // 1152, K ordered as tap*128 + ic
#define PIX   (RES * RES)     // 65536
#define SCOLS 66              // 64-pixel tile + 2 halo columns

typedef __bf16 v16bf __attribute__((ext_vector_type(16)));
typedef float  v8f   __attribute__((ext_vector_type(8)));

union Frag {
    v16bf v;
    uint4 q[2];
};

union Pack8 {
    __bf16 h[8];
    uint4  q;
};

static __device__ __forceinline__ unsigned short f32_to_bf16(float f) {
    union { float f; unsigned int u; } v;
    v.f = f;
    unsigned int u = v.u;
    u += 0x7FFFu + ((u >> 16) & 1u);   // round-to-nearest-even
    return (unsigned short)(u >> 16);
}

static __device__ __forceinline__ float wave_reduce_add(float v) {
    #pragma unroll
    for (int off = 16; off > 0; off >>= 1)
        v += __shfl_xor(v, off, 32);
    return v;
}

// ---------------------------------------------------------------------------
// Kernel 1: styles[n][ic] = dot(w[n,:], affine_w[ic,:]) / sqrt(WDIM) + affine_b[ic]
// ---------------------------------------------------------------------------
__global__ __launch_bounds__(256)
void styles_kernel(const float* __restrict__ w, const float* __restrict__ aff_w,
                   const float* __restrict__ aff_b, float* __restrict__ styles) {
    const int wid  = blockIdx.x * 8 + (threadIdx.x >> 5);   // 0..1023
    const int lane = threadIdx.x & 31;
    const int n  = wid >> 7;
    const int ic = wid & 127;

    float sum = 0.f;
    #pragma unroll
    for (int k = 0; k < WDIM; k += 32)
        sum += w[n * WDIM + k + lane] * aff_w[ic * WDIM + k + lane];
    sum = wave_reduce_add(sum);
    if (lane == 0)
        styles[n * IC + ic] = sum * 0.044194173824159216f /*1/sqrt(512)*/ + aff_b[ic];
}

// ---------------------------------------------------------------------------
// Kernel 2: demodulated bf16 weights, permuted to [n][oc][tap*128 + ic].
// ---------------------------------------------------------------------------
__global__ __launch_bounds__(256)
void modw_kernel(const float* __restrict__ weight, const float* __restrict__ styles,
                 unsigned short* __restrict__ wmod) {
    const int wid  = blockIdx.x * 8 + (threadIdx.x >> 5);   // 0..1023
    const int lane = threadIdx.x & 31;
    const int n  = wid >> 7;
    const int oc = wid & 127;

    const float* wrow = weight + (size_t)oc * KTOT;         // [ic*9 + tap] layout
    const float* srow = styles + n * IC;

    float ssq = 0.f;
    #pragma unroll
    for (int e = lane; e < KTOT; e += 32) {
        const float v = wrow[e] * srow[e / 9];
        ssq += v * v;
    }
    ssq = wave_reduce_add(ssq);
    const float d = rsqrtf(ssq + 1e-8f);

    unsigned short* dst = wmod + (size_t)(n * OC + oc) * KTOT;
    #pragma unroll
    for (int e = lane; e < KTOT; e += 32) {
        const int ic  = e / 9;
        const int tap = e - ic * 9;
        const float v = wrow[e] * srow[ic] * d;
        dst[tap * IC + ic] = f32_to_bf16(v);
    }
}

// ---------------------------------------------------------------------------
// Kernel 3: implicit-GEMM conv + epilogue.
// Block = 128 threads (4 waves). Tile: 64 pixels x 128 OC.
// Wave wv owns OC stripe [wv*32, wv*32+32) -> 8 accumulators (4 px-groups x 2).
// ---------------------------------------------------------------------------
__global__ __launch_bounds__(128)
void conv_kernel(const float* __restrict__ x, const unsigned short* __restrict__ wmod,
                 const float* __restrict__ noise_const,
                 const float* __restrict__ noise_strength,
                 const float* __restrict__ bias, float* __restrict__ out) {
    // x halo slab: [3 rows][66 cols][128 ic] bf16 = 50,688 B
    __shared__ __align__(16) unsigned short Slab[3 * SCOLS * IC];

    const int n     = blockIdx.y;
    const int pbase = blockIdx.x * 64;           // 64 consecutive pixels, one row
    const int py    = pbase >> 8;
    const int pxb   = pbase & 255;               // 0,64,128,192

    const int t      = threadIdx.x;              // 0..127
    const int lane   = t & 31;
    const int wv     = t >> 5;                   // wave 0..3
    const int laneLo = lane & 15;
    const int laneHi = lane >> 4;

    const float* xb = x + (size_t)n * IC * PIX;

    // ---------------- stage halo slab (fp32 -> bf16, packed b128 stores) ----
    {
        const int col  = t & 63;                 // slab column 0..63
        const int part = t >> 6;                 // ic half: 0 or 1
        const int ix   = pxb - 1 + col;
        const bool vc  = (ix >= 0);              // ix < 256 always here
        #pragma unroll
        for (int row = 0; row < 3; ++row) {
            const int iy  = py - 1 + row;
            const bool ok = vc && (iy >= 0) && (iy < RES);
            const long long boff = ok ? ((long long)iy * RES + ix) : 0;
            const float* xp = xb + (long long)(part * 64) * PIX + boff;
            unsigned short* dst = Slab + (row * SCOLS + col) * IC + part * 64;
            #pragma unroll
            for (int v8 = 0; v8 < 64; v8 += 8) {
                Pack8 pk;
                #pragma unroll
                for (int q = 0; q < 8; ++q) {
                    const float f = ok ? xp[(long long)(v8 + q) * PIX] : 0.0f;
                    pk.h[q] = (__bf16)f;
                }
                *(uint4*)(dst + v8) = pk.q;
            }
        }
        // halo columns 64,65: 3*2*128 = 768 elements, 6 per thread
        #pragma unroll
        for (int k = 0; k < 6; ++k) {
            const int e    = t + k * 128;        // 0..767
            const int row  = e >> 8;
            const int rem  = e & 255;
            const int colx = 64 + (rem >> 7);
            const int ic   = rem & 127;
            const int iy   = py - 1 + row;
            const int ix2  = pxb - 1 + colx;
            const bool ok  = (iy >= 0) && (iy < RES) && (ix2 >= 0) && (ix2 < RES);
            const float f  = ok ? xb[(size_t)ic * PIX + iy * RES + ix2] : 0.0f;
            Slab[(row * SCOLS + colx) * IC + ic] = f32_to_bf16(f);
        }
    }
    __syncthreads();

    // ---------------- main loop: barrier-free, WMMA-dominant ----------------
    // A frag base: pixel row m = laneLo, K halves per documented 16-bit layout
    const unsigned short* aBase = Slab + laneLo * IC + laneHi * 8;
    // B frag base: column oc = wv*32 + j*16 + laneLo, K halves per documented 16-bit layout
    const unsigned short* bBase =
        wmod + ((size_t)(n * OC) + wv * 32 + laneLo) * KTOT + laneHi * 8;

    v8f acc[4][2] = {};

    #pragma unroll
    for (int tap = 0; tap < 9; ++tap) {
        const int ky = tap / 3;
        const int kx = tap - ky * 3;
        #pragma unroll
        for (int c = 0; c < 4; ++c) {
            const int kb = tap * 128 + c * 32;

            Frag b[2];
            #pragma unroll
            for (int j = 0; j < 2; ++j) {
                const unsigned short* bp = bBase + (size_t)j * 16 * KTOT + kb;
                b[j].q[0] = *(const uint4*)(bp);
                b[j].q[1] = *(const uint4*)(bp + 16);
            }

            Frag a[4];
            #pragma unroll
            for (int i = 0; i < 4; ++i) {
                const unsigned short* ap =
                    aBase + ((ky * SCOLS + kx) * IC) + i * (16 * IC) + c * 32;
                a[i].q[0] = *(const uint4*)(ap);        // K 0-7 / 8-15
                a[i].q[1] = *(const uint4*)(ap + 16);   // K 16-23 / 24-31
            }

            #pragma unroll
            for (int i = 0; i < 4; ++i)
                #pragma unroll
                for (int j = 0; j < 2; ++j)
                    acc[i][j] = __builtin_amdgcn_wmma_f32_16x16x32_bf16(
                        false, a[i].v, false, b[j].v, (short)0, acc[i][j],
                        false, false);
        }
    }

    // ---------------- epilogue: noise + bias + lrelu*sqrt(2) + clamp -------
    const float strength = noise_strength[0];
    #pragma unroll
    for (int j = 0; j < 2; ++j) {
        const int oc = wv * 32 + j * 16 + laneLo;
        const float bv = bias[oc];
        float* orow = out + ((size_t)(n * OC + oc) << 16);
        #pragma unroll
        for (int i = 0; i < 4; ++i) {
            const int p0 = pbase + i * 16 + laneHi * 8;
            #pragma unroll
            for (int r = 0; r < 8; ++r) {
                float y = acc[i][j][r] + noise_const[p0 + r] * strength + bv;
                y = (y >= 0.0f ? y : 0.2f * y) * 1.4142135623730951f;
                y = fminf(fmaxf(y, -256.0f), 256.0f);
                orow[p0 + r] = y;
            }
        }
    }
}

// ---------------------------------------------------------------------------
extern "C" void kernel_launch(void* const* d_in, const int* in_sizes, int n_in,
                              void* d_out, int out_size, void* d_ws, size_t ws_size,
                              hipStream_t stream) {
    const float* x              = (const float*)d_in[0];
    const float* w              = (const float*)d_in[1];
    const float* affine_w       = (const float*)d_in[2];
    const float* affine_b       = (const float*)d_in[3];
    const float* weight         = (const float*)d_in[4];
    const float* bias           = (const float*)d_in[5];
    const float* noise_const    = (const float*)d_in[6];
    const float* noise_strength = (const float*)d_in[7];
    float* out = (float*)d_out;

    // workspace: styles fp32 (4 KB) | wmod bf16 [8][128][1152] (2.25 MB)
    float* styles        = (float*)d_ws;
    unsigned short* wmod = (unsigned short*)((char*)d_ws + 4096);

    styles_kernel<<<128, 256, 0, stream>>>(w, affine_w, affine_b, styles);
    modw_kernel<<<128, 256, 0, stream>>>(weight, styles, wmod);

    // conv: 1024 pixel-tiles x 8 samples, 128 threads (4 waves) per block
    dim3 grid(PIX / 64, N_SMP);
    conv_kernel<<<grid, 128, 0, stream>>>(x, wmod, noise_const, noise_strength, bias, out);
}
